// TKGEncoder_33938831573718
// MI455X (gfx1250) — matrix operand order repaired and checked by reading, not compiled, not run on hardware
//
#include <hip/hip_runtime.h>

// ---------------- problem constants (match reference) ----------------
#define HEADS   4
#define CC      128      // channels
#define DD      32       // per-head dim
#define LLAY    2        // HGT layers
#define TKG_LEN 24
#define NU      2000
#define NL      5000
#define EE      40000
#define BB      64
#define SS      128

#define ROWS_PER_BLK 64
#define MTILES 4         // 4 x 16-row WMMA tiles per block

typedef __attribute__((ext_vector_type(16))) _Float16 v16h;
typedef __attribute__((ext_vector_type(8)))  float    v8f;

__device__ __forceinline__ float gelu_exact(float x) {
  return 0.5f * x * (1.0f + erff(x * 0.70710678118654752440f));
}
// order-preserving float<->uint encoding for atomic segment-max
__device__ __forceinline__ unsigned f2ord(float f) {
  unsigned u = __float_as_uint(f);
  return (u & 0x80000000u) ? ~u : (u | 0x80000000u);
}
__device__ __forceinline__ float ord2f(unsigned u) {
  unsigned b = (u & 0x80000000u) ? (u & 0x7FFFFFFFu) : ~u;
  return __uint_as_float(b);
}

// ---------------------------------------------------------------------
// Fused GEMM: Out[M,128] = epilogue( act(A[gather])[M,128] @ W[128,128] + bias )
// mode 0: relu(acc+b)   mode 1: acc+b   mode 2: g*(acc+b)+(1-g)*Xskip
// 64 rows per block (4 WMMA M-tiles); 8 waves each own one 16-col N-tile.
// W staged TRANSPOSED in LDS so B-fragments are contiguous b128 loads, and
// each B-fragment is loaded once per K-step and reused across 4 WMMAs.
// ---------------------------------------------------------------------
__global__ __launch_bounds__(256) void k_gemm128(
    const float* __restrict__ A, const int* __restrict__ ids,
    const float* __restrict__ W, const float* __restrict__ bias,
    const float* __restrict__ Xskip, const float* __restrict__ gatePtr,
    float* __restrict__ Out, int M, int actGelu, int mode)
{
  __shared__ _Float16 sA[ROWS_PER_BLK][CC + 8];   // row-major activations
  __shared__ _Float16 sWt[CC][CC + 8];            // sWt[n][k] = W[k][n]
  const int tid = threadIdx.x;
  const int rowBase = blockIdx.x * ROWS_PER_BLK;

  for (int i = tid; i < ROWS_PER_BLK * CC; i += 256) {
    int r = i >> 7, c = i & (CC - 1);
    int row = rowBase + r;
    float v = 0.0f;
    if (row < M) {
      int g = ids ? ids[row] : row;
      v = A[(size_t)g * CC + c];
      if (actGelu) v = gelu_exact(v);
    }
    sA[r][c] = (_Float16)v;
  }
  for (int i = tid; i < CC * CC; i += 256) {
    int k = i >> 7, n = i & (CC - 1);
    sWt[n][k] = (_Float16)W[i];
  }
  __syncthreads();

  const int wave = tid >> 5;        // N-tile index 0..7
  const int lane = tid & 31;
  const int half = lane >> 4;       // lane group
  const int l16  = lane & 15;       // A: row m ; B: col n

  v8f acc[MTILES] = {{}, {}, {}, {}};
#pragma unroll
  for (int kb = 0; kb < CC; kb += 32) {
    // B fragment: lanes 0-15 K=kb..kb+15, lanes 16-31 K=kb+16..kb+31 (contiguous)
    v16h bf;
#pragma unroll
    for (int j = 0; j < 16; ++j)
      bf[j] = sWt[wave * 16 + l16][kb + half * 16 + j];
#pragma unroll
    for (int mt = 0; mt < MTILES; ++mt) {
      // A fragment: lanes 0-15 K=kb..kb+15, lanes 16-31 K=kb+16..kb+31 (contiguous)
      v16h af;
#pragma unroll
      for (int j = 0; j < 16; ++j) {
        int ka = kb + half * 16 + j;
        af[j] = sA[mt * 16 + l16][ka];
      }
      acc[mt] = __builtin_amdgcn_wmma_f32_16x16x32_f16(false, af, false, bf,
                                                       (short)0, acc[mt], false, false);
    }
  }

  float g = 0.0f;
  if (mode == 2) g = 1.0f / (1.0f + expf(-gatePtr[0]));
  const int col = wave * 16 + l16;
  const float bv = bias ? bias[col] : 0.0f;
#pragma unroll
  for (int mt = 0; mt < MTILES; ++mt) {
#pragma unroll
    for (int r = 0; r < 8; ++r) {
      int row = rowBase + mt * 16 + r + half * 8;   // lanes 16-31 hold rows 8..15
      if (row < M) {
        float v = acc[mt][r] + bv;
        if (mode == 0) v = fmaxf(v, 0.0f);
        else if (mode == 2) v = g * v + (1.0f - g) * Xskip[(size_t)row * CC + col];
        Out[(size_t)row * CC + col] = v;
      }
    }
  }
}

// ---------------------------------------------------------------------
// Per-head relation transform: Out[n, h*32+e] = sum_d X[n, h*32+d] * R[h][d][e]
// 64 rows per block; 8 waves = 4 heads x 2 sub-tiles, each loops 4 M-tiles
// reusing its B fragment. R staged transposed: sRt[h][e][d].
// ---------------------------------------------------------------------
__global__ __launch_bounds__(256) void k_relxform(
    const float* __restrict__ X, const float* __restrict__ R,   // R: [4][32][32]
    float* __restrict__ Out, int M)
{
  __shared__ _Float16 sX[ROWS_PER_BLK][CC + 8];
  __shared__ _Float16 sRt[HEADS][DD][DD + 8];
  const int tid = threadIdx.x;
  const int rowBase = blockIdx.x * ROWS_PER_BLK;

  for (int i = tid; i < ROWS_PER_BLK * CC; i += 256) {
    int r = i >> 7, c = i & (CC - 1);
    int row = rowBase + r;
    sX[r][c] = (_Float16)(row < M ? X[(size_t)row * CC + c] : 0.0f);
  }
  for (int i = tid; i < HEADS * DD * DD; i += 256) {
    int h = i >> 10, d = (i >> 5) & 31, e = i & 31;
    sRt[h][e][d] = (_Float16)R[i];
  }
  __syncthreads();

  const int wave = tid >> 5;
  const int head = wave >> 1, sub = wave & 1;
  const int lane = tid & 31, half = lane >> 4, l16 = lane & 15;

  v16h bf;
#pragma unroll
  for (int j = 0; j < 16; ++j)
    bf[j] = sRt[head][sub * 16 + l16][half * 16 + j];

  v8f acc[MTILES] = {{}, {}, {}, {}};
#pragma unroll
  for (int mt = 0; mt < MTILES; ++mt) {
    v16h af;
#pragma unroll
    for (int j = 0; j < 16; ++j) {
      int ka = half * 16 + j;        // K within head, 0..31
      af[j] = sX[mt * 16 + l16][head * DD + ka];
    }
    acc[mt] = __builtin_amdgcn_wmma_f32_16x16x32_f16(false, af, false, bf,
                                                     (short)0, acc[mt], false, false);
  }

  const int col = head * DD + sub * 16 + l16;
#pragma unroll
  for (int mt = 0; mt < MTILES; ++mt) {
#pragma unroll
    for (int r = 0; r < 8; ++r) {
      int row = rowBase + mt * 16 + r + half * 8;
      if (row < M) Out[(size_t)row * CC + col] = acc[mt][r];
    }
  }
}

// ------------------------- edge-attention kernels --------------------
__global__ __launch_bounds__(256) void k_edge_logits(
    const float* __restrict__ q, const float* __restrict__ kr,
    const int* __restrict__ src, const int* __restrict__ dst,
    const float* __restrict__ p, float* __restrict__ ebuf,
    unsigned* __restrict__ smax)
{
  int t = blockIdx.x * 256 + threadIdx.x;
  if (t >= EE * HEADS) return;
  int e = t >> 2, h = t & 3;
  int s = src[e], d = dst[e];
  const float4* qv = (const float4*)(q  + (size_t)d * CC + h * DD);
  const float4* kv = (const float4*)(kr + (size_t)s * CC + h * DD);
  float acc = 0.0f;
#pragma unroll
  for (int i = 0; i < 8; ++i) {
    float4 a = qv[i], b = kv[i];
    acc = fmaf(a.x, b.x, acc); acc = fmaf(a.y, b.y, acc);
    acc = fmaf(a.z, b.z, acc); acc = fmaf(a.w, b.w, acc);
  }
  float logit = acc * p[h] * 0.17677669529663687f;  // 1/sqrt(32)
  ebuf[t] = logit;
  atomicMax(&smax[d * HEADS + h], f2ord(logit));
}

__global__ __launch_bounds__(256) void k_edge_expsum(
    const int* __restrict__ dst, const unsigned* __restrict__ smax,
    float* __restrict__ ebuf, float* __restrict__ ssum)
{
  int t = blockIdx.x * 256 + threadIdx.x;
  if (t >= EE * HEADS) return;
  int e = t >> 2, h = t & 3;
  int d = dst[e];
  float m = ord2f(smax[d * HEADS + h]);
  float ev = expf(ebuf[t] - m);
  ebuf[t] = ev;
  atomicAdd(&ssum[d * HEADS + h], ev);
}

__global__ __launch_bounds__(256) void k_edge_agg(
    const float* __restrict__ ebuf, const float* __restrict__ mr,
    const int* __restrict__ src, const int* __restrict__ dst,
    const float* __restrict__ ssum, float* __restrict__ agg)
{
  int t = blockIdx.x * 256 + threadIdx.x;
  if (t >= EE * CC) return;
  int e = t >> 7, c = t & (CC - 1), h = c >> 5;
  int s = src[e], d = dst[e];
  float alpha = ebuf[e * HEADS + h] / (ssum[d * HEADS + h] + 1e-16f);
  atomicAdd(&agg[(size_t)d * CC + c], mr[(size_t)s * CC + c] * alpha);
}

// ----------------------------- utilities -----------------------------
__global__ __launch_bounds__(256) void k_fill_f32(float* p, float v, int n) {
  int t = blockIdx.x * 256 + threadIdx.x;
  if (t < n) p[t] = v;
}
__global__ __launch_bounds__(256) void k_fill_u32(unsigned* p, unsigned v, int n) {
  int t = blockIdx.x * 256 + threadIdx.x;
  if (t < n) p[t] = v;
}
__global__ __launch_bounds__(256) void k_gather_traj(
    const float* __restrict__ tkg_all, const int* __restrict__ tkg_idx,
    const int* __restrict__ traj, float* __restrict__ out)
{
  int t = blockIdx.x * 256 + threadIdx.x;
  if (t >= BB * SS * CC) return;
  int c = t & (CC - 1);
  int bs = t >> 7;
  int ti = tkg_idx[bs];
  int lo = traj[bs];
  out[t] = tkg_all[((size_t)ti * NL + lo) * CC + c];
}

// ---------------------------------------------------------------------
extern "C" void kernel_launch(void* const* d_in, const int* in_sizes, int n_in,
                              void* d_out, int out_size, void* d_ws, size_t ws_size,
                              hipStream_t stream) {
  (void)in_sizes; (void)n_in; (void)out_size; (void)ws_size;

  const float* user_emb   = (const float*)d_in[0];
  const float* loc_emb    = (const float*)d_in[1];
  const float* lin_user_w = (const float*)d_in[2];
  const float* lin_user_b = (const float*)d_in[3];
  const float* lin_loc_w  = (const float*)d_in[4];
  const float* lin_loc_b  = (const float*)d_in[5];
  const float* w_user     = (const float*)d_in[6];
  const float* b_user     = (const float*)d_in[7];
  const float* w_loc      = (const float*)d_in[8];
  const float* b_loc      = (const float*)d_in[9];
  const float* rel_a      = (const float*)d_in[10];
  const float* rel_m      = (const float*)d_in[11];
  const float* rel_p      = (const float*)d_in[12];
  const float* skip       = (const float*)d_in[13];
  const float* lin2_w     = (const float*)d_in[14];
  const float* lin2_b     = (const float*)d_in[15];
  const int* user_ids     = (const int*)d_in[16];
  const int* loc_ids      = (const int*)d_in[17];
  const int* ei_ul_src    = (const int*)d_in[18];
  const int* ei_ul_dst    = (const int*)d_in[19];
  const int* ei_lu_src    = (const int*)d_in[20];
  const int* ei_lu_dst    = (const int*)d_in[21];
  const int* traj         = (const int*)d_in[22];
  const int* tkg_idx      = (const int*)d_in[23];

  float* traj_out = (float*)d_out;                              // [B,S,C]
  float* tkg_all  = (float*)d_out + (size_t)BB * SS * CC;       // [T,NL,C]

  // workspace layout (~30.2 MB)
  float* wsf = (float*)d_ws;
  size_t o = 0;
  float* xu  = wsf + o; o += (size_t)NU * CC;
  float* xl  = wsf + o; o += (size_t)NL * CC;
  float* ku  = wsf + o; o += (size_t)NU * CC;
  float* qu  = wsf + o; o += (size_t)NU * CC;
  float* vu  = wsf + o; o += (size_t)NU * CC;
  float* kl  = wsf + o; o += (size_t)NL * CC;
  float* ql  = wsf + o; o += (size_t)NL * CC;
  float* vl  = wsf + o; o += (size_t)NL * CC;
  float* kru = wsf + o; o += (size_t)NU * CC;
  float* mru = wsf + o; o += (size_t)NU * CC;
  float* krl = wsf + o; o += (size_t)NL * CC;
  float* mrl = wsf + o; o += (size_t)NL * CC;
  float* ebuf_ul = wsf + o; o += (size_t)EE * HEADS;
  float* ebuf_lu = wsf + o; o += (size_t)EE * HEADS;
  unsigned* smax_l = (unsigned*)(wsf + o); o += (size_t)NL * HEADS;
  float* ssum_l = wsf + o; o += (size_t)NL * HEADS;
  unsigned* smax_u = (unsigned*)(wsf + o); o += (size_t)NU * HEADS;
  float* ssum_u = wsf + o; o += (size_t)NU * HEADS;
  float* agg_l = wsf + o; o += (size_t)NL * CC;
  float* agg_u = wsf + o; o += (size_t)NU * CC;
  float* xu2 = wsf + o; o += (size_t)NU * CC;
  float* xl2 = wsf + o; o += (size_t)NL * CC;

  const int GU  = (NU + ROWS_PER_BLK - 1) / ROWS_PER_BLK;   // 32
  const int GL  = (NL + ROWS_PER_BLK - 1) / ROWS_PER_BLK;   // 79
  const int GE4 = (EE * HEADS + 255) / 256;                 // 625
  const int GEC = (EE * CC + 255) / 256;                    // 20000
  const int GLH = (NL * HEADS + 255) / 256;
  const int GUH = (NU * HEADS + 255) / 256;
  const int GAL = (NL * CC + 255) / 256;
  const int GAU = (NU * CC + 255) / 256;
  const int GOT = (BB * SS * CC + 255) / 256;

  for (int t = 0; t < TKG_LEN; ++t) {
    const int* uid = user_ids + (size_t)t * NU;
    const int* lid = loc_ids  + (size_t)t * NL;
    const int* sul = ei_ul_src + (size_t)t * EE;
    const int* dul = ei_ul_dst + (size_t)t * EE;
    const int* slu = ei_lu_src + (size_t)t * EE;
    const int* dlu = ei_lu_dst + (size_t)t * EE;

    // embedding lookup + per-type linear + relu
    k_gemm128<<<GU, 256, 0, stream>>>(user_emb, uid, lin_user_w, lin_user_b,
                                      nullptr, nullptr, xu, NU, 0, 0);
    k_gemm128<<<GL, 256, 0, stream>>>(loc_emb, lid, lin_loc_w, lin_loc_b,
                                      nullptr, nullptr, xl, NL, 0, 0);

    float* cxu = xu; float* cxl = xl;
    float* nxu = xu2; float* nxl = xl2;

    for (int l = 0; l < LLAY; ++l) {
      const float* wu = w_user + (size_t)l * 4 * CC * CC;
      const float* bu = b_user + (size_t)l * 4 * CC;
      const float* wl = w_loc  + (size_t)l * 4 * CC * CC;
      const float* bl = b_loc  + (size_t)l * 4 * CC;
      const float* ra = rel_a  + (size_t)l * 2 * HEADS * DD * DD;
      const float* rm = rel_m  + (size_t)l * 2 * HEADS * DD * DD;
      const float* rp = rel_p  + (size_t)l * 2 * HEADS;
      const float* sk = skip   + (size_t)l * 2;

      // k, q, v per node type
      k_gemm128<<<GU, 256, 0, stream>>>(cxu, nullptr, wu + 0 * CC * CC, bu + 0 * CC, nullptr, nullptr, ku, NU, 0, 1);
      k_gemm128<<<GU, 256, 0, stream>>>(cxu, nullptr, wu + 1 * CC * CC, bu + 1 * CC, nullptr, nullptr, qu, NU, 0, 1);
      k_gemm128<<<GU, 256, 0, stream>>>(cxu, nullptr, wu + 2 * CC * CC, bu + 2 * CC, nullptr, nullptr, vu, NU, 0, 1);
      k_gemm128<<<GL, 256, 0, stream>>>(cxl, nullptr, wl + 0 * CC * CC, bl + 0 * CC, nullptr, nullptr, kl, NL, 0, 1);
      k_gemm128<<<GL, 256, 0, stream>>>(cxl, nullptr, wl + 1 * CC * CC, bl + 1 * CC, nullptr, nullptr, ql, NL, 0, 1);
      k_gemm128<<<GL, 256, 0, stream>>>(cxl, nullptr, wl + 2 * CC * CC, bl + 2 * CC, nullptr, nullptr, vl, NL, 0, 1);

      // per-head relation transforms: rel 0 = user->loc (on user k/v), rel 1 = loc->user
      k_relxform<<<GU, 256, 0, stream>>>(ku, ra, kru, NU);
      k_relxform<<<GU, 256, 0, stream>>>(vu, rm, mru, NU);
      k_relxform<<<GL, 256, 0, stream>>>(kl, ra + (size_t)HEADS * DD * DD, krl, NL);
      k_relxform<<<GL, 256, 0, stream>>>(vl, rm + (size_t)HEADS * DD * DD, mrl, NL);

      // reset segment state + accumulators
      k_fill_u32<<<GLH, 256, 0, stream>>>(smax_l, 0u, NL * HEADS);
      k_fill_f32<<<GLH, 256, 0, stream>>>(ssum_l, 0.0f, NL * HEADS);
      k_fill_f32<<<GAL, 256, 0, stream>>>(agg_l, 0.0f, NL * CC);
      k_fill_u32<<<GUH, 256, 0, stream>>>(smax_u, 0u, NU * HEADS);
      k_fill_f32<<<GUH, 256, 0, stream>>>(ssum_u, 0.0f, NU * HEADS);
      k_fill_f32<<<GAU, 256, 0, stream>>>(agg_u, 0.0f, NU * CC);

      // edge type 0: user -> loc (dst = locations, q = ql)
      k_edge_logits<<<GE4, 256, 0, stream>>>(ql, kru, sul, dul, rp, ebuf_ul, smax_l);
      k_edge_expsum<<<GE4, 256, 0, stream>>>(dul, smax_l, ebuf_ul, ssum_l);
      k_edge_agg   <<<GEC, 256, 0, stream>>>(ebuf_ul, mru, sul, dul, ssum_l, agg_l);

      // edge type 1: loc -> user (dst = users, q = qu)
      k_edge_logits<<<GE4, 256, 0, stream>>>(qu, krl, slu, dlu, rp + HEADS, ebuf_lu, smax_u);
      k_edge_expsum<<<GE4, 256, 0, stream>>>(dlu, smax_u, ebuf_lu, ssum_u);
      k_edge_agg   <<<GEC, 256, 0, stream>>>(ebuf_lu, mrl, slu, dlu, ssum_u, agg_u);

      // gated GELU update: x' = g*(gelu(agg)@Wa + ba) + (1-g)*x
      k_gemm128<<<GU, 256, 0, stream>>>(agg_u, nullptr, wu + 3 * CC * CC, bu + 3 * CC, cxu, sk + 0, nxu, NU, 1, 2);
      k_gemm128<<<GL, 256, 0, stream>>>(agg_l, nullptr, wl + 3 * CC * CC, bl + 3 * CC, cxl, sk + 1, nxl, NL, 1, 2);

      float* tmp;
      tmp = cxu; cxu = nxu; nxu = tmp;
      tmp = cxl; cxl = nxl; nxl = tmp;
    }

    // lin2 on locations -> tkg_out[t]
    k_gemm128<<<GL, 256, 0, stream>>>(cxl, nullptr, lin2_w, lin2_b, nullptr, nullptr,
                                      tkg_all + (size_t)t * NL * CC, NL, 0, 1);
  }

  // tkg_traj[b,s,:] = tkg_out[tkg_idx[b,s], traj[b,s], :]
  k_gather_traj<<<GOT, 256, 0, stream>>>(tkg_all, tkg_idx, traj, traj_out);
}
